// Attention_6562710028374
// MI455X (gfx1250) — matrix operand order repaired and checked by reading, not compiled, not run on hardware
//
#include <hip/hip_runtime.h>
#include <hip/hip_bf16.h>
#include <math.h>

// ---------------- problem constants (fixed by the reference) ----------------
#define DIMX   1024
#define DHEAD  64
#define NHEADS 16
#define BATCH  2
#define SEQ    2048
#define ROWS   (BATCH * SEQ)   // 4096
#define LN_EPSF  1e-5f
#define RMS_EPSF 1e-8f

typedef __attribute__((ext_vector_type(16))) _Float16 v16h;
typedef __attribute__((ext_vector_type(8)))  _Float16 v8h;
typedef __attribute__((ext_vector_type(4)))  _Float16 v4h;
typedef __attribute__((ext_vector_type(8)))  float    v8f;

union V16U { v16h v; v8h h[2]; };

// D = A(16x32 f16) * B(32x16 f16) + C(16x16 f32)
__device__ inline v8f wmma32(v16h a, v16h b, v8f c) {
    return __builtin_amdgcn_wmma_f32_16x16x32_f16(
        /*neg_a=*/false, a, /*neg_b=*/false, b,
        /*c_mod=*/(short)0, c, /*reuse_a=*/false, /*reuse_b=*/false);
}

// A operand, 16x32 (MxK), row-major source with leading dim lda (elements).
// ISA layout: lane<16 holds row m=lane, halves {K:g*8..g*8+7, K:16+g*8..+7}.
__device__ inline v16h load_a_nt(const _Float16* __restrict__ A, int lda,
                                 int m0, int k0) {
    const int lane = threadIdx.x & 31;
    const int m = lane & 15, g = lane >> 4;
    const _Float16* p = A + (size_t)(m0 + m) * lda + k0 + g * 8;
    V16U r;
    r.h[0] = *(const v8h*)(p);
    r.h[1] = *(const v8h*)(p + 16);
    return r.v;
}

// B operand, 32x16 (KxN), where "row n" of Bm is contiguous along K
// (NT gemm: Bm[n*ldb + k]).  Lane group g holds K = g*8..g*8+7, 16+g*8..+7.
__device__ inline v16h load_b_nt(const _Float16* __restrict__ Bm, int ldb,
                                 int n0, int k0) {
    const int lane = threadIdx.x & 31;
    const int n = lane & 15, g = lane >> 4;
    const _Float16* p = Bm + (size_t)(n0 + n) * ldb + k0 + g * 8;
    V16U r;
    r.h[0] = *(const v8h*)(p);
    r.h[1] = *(const v8h*)(p + 16);
    return r.v;
}

// intra-wave LDS ordering: CDNA5 split counter (cheaper than a block barrier;
// LDS region here is private per wave, so no cross-wave sync is needed)
__device__ inline void lds_fence_wave() {
    asm volatile("s_wait_dscnt 0x0" ::: "memory");
}

// ---------------- fp32 -> f16 conversion (4 elems/thread) ----------------
__global__ __launch_bounds__(256) void cvt_f16_kernel(
        const float* __restrict__ s, _Float16* __restrict__ d, int n4) {
    int i = blockIdx.x * 256 + threadIdx.x;
    if (i < n4) {
        float4 v = ((const float4*)s)[i];
        v4h o = { (_Float16)v.x, (_Float16)v.y, (_Float16)v.z, (_Float16)v.w };
        ((v4h*)d)[i] = o;
    }
}

// ---------------- LayerNorm: x (4096x1024) -> xn f16 ----------------
__global__ __launch_bounds__(256) void ln_kernel(
        const float* __restrict__ x, const float* __restrict__ gamma,
        const float* __restrict__ beta, _Float16* __restrict__ xn) {
    const int row = blockIdx.x;
    const int tid = threadIdx.x;
    const float* xr = x + (size_t)row * DIMX;
    const float4 xv = ((const float4*)xr)[tid];          // 4 elems/thread
    float s  = xv.x + xv.y + xv.z + xv.w;
    float s2 = xv.x * xv.x + xv.y * xv.y + xv.z * xv.z + xv.w * xv.w;
    __shared__ float rs[256], rs2[256];
    rs[tid] = s; rs2[tid] = s2;
    __syncthreads();
    for (int st = 128; st > 0; st >>= 1) {
        if (tid < st) { rs[tid] += rs[tid + st]; rs2[tid] += rs2[tid + st]; }
        __syncthreads();
    }
    const float mean = rs[0] * (1.0f / DIMX);
    const float var  = rs2[0] * (1.0f / DIMX) - mean * mean;
    const float inv  = rsqrtf(var + LN_EPSF);
    const float4 gv = ((const float4*)gamma)[tid];
    const float4 bv = ((const float4*)beta)[tid];
    v4h o = { (_Float16)((xv.x - mean) * inv * gv.x + bv.x),
              (_Float16)((xv.y - mean) * inv * gv.y + bv.y),
              (_Float16)((xv.z - mean) * inv * gv.z + bv.z),
              (_Float16)((xv.w - mean) * inv * gv.w + bv.w) };
    ((v4h*)(xn + (size_t)row * DIMX))[tid] = o;
}

// ---------------- NT GEMM, 4 waves/block, 16x64 tile per wave ---------------
// Waves in a block share the same 64-col W tile => B loads hit WGP$/L1.
// C[i,j] = sum_k A[i,k] * W[j,k]
// mode 0: V projection  -> dst16 as vT (B,H,Dh,N) f16
// mode 1: Q projection  -> RMSNorm * dh^-0.5 * q_gamma -> qn (B,H,N,Dh) f16
// mode 2: K projection  -> RMSNorm * k_gamma           -> kn (B,H,N,Dh) f16
// mode 3: out proj      -> fp32 + bias                 -> out (B,N,DIMX)
__global__ __launch_bounds__(128) void gemm_nt_kernel(
        const _Float16* __restrict__ A, const _Float16* __restrict__ W,
        _Float16* __restrict__ dst16, float* __restrict__ dst32,
        const float* __restrict__ aux, int mode) {
    const int lane = threadIdx.x & 31;
    const int wv   = threadIdx.x >> 5;            // wave 0..3
    const int ln = lane & 15, lg = lane >> 4;
    const int m0 = blockIdx.x * 64 + wv * 16;     // output row block
    const int n0 = blockIdx.y * 64;               // output col block (1 head)
    const int h  = blockIdx.y;

    v8f acc[4] = {};
    for (int k0 = 0; k0 < DIMX; k0 += 32) {
        v16h a = load_a_nt(A, DIMX, m0, k0);
        #pragma unroll
        for (int t = 0; t < 4; ++t) {
            v16h b = load_b_nt(W, DIMX, n0 + t * 16, k0);
            acc[t] = wmma32(a, b, acc[t]);
        }
    }

    if (mode == 1 || mode == 2) {
        // fused per-row RMSNorm over the 64-wide head tile
        float ss[8];
        #pragma unroll
        for (int vi = 0; vi < 8; ++vi) {
            float p = 0.f;
            #pragma unroll
            for (int t = 0; t < 4; ++t) p += acc[t][vi] * acc[t][vi];
            p += __shfl_xor(p, 1, 32);
            p += __shfl_xor(p, 2, 32);
            p += __shfl_xor(p, 4, 32);
            p += __shfl_xor(p, 8, 32);
            ss[vi] = p;
        }
        const float extra = (mode == 1) ? 0.125f : 1.0f;   // dh^-0.5 for Q
        #pragma unroll
        for (int vi = 0; vi < 8; ++vi) {
            const float nrm = sqrtf(ss[vi] * (1.0f / 64.0f));
            const float sc  = extra / fmaxf(nrm, RMS_EPSF);
            const int r = lg * 8 + vi;
            const int i = m0 + r;
            const int b = i >> 11, n = i & (SEQ - 1);
            #pragma unroll
            for (int t = 0; t < 4; ++t) {
                const int d = t * 16 + ln;
                const float val = acc[t][vi] * sc * aux[d];
                dst16[((size_t)(b * NHEADS + h) * SEQ + n) * DHEAD + d] =
                    (_Float16)val;
            }
        }
    } else if (mode == 0) {
        // V: store transposed per head -> vT[(b,h,d,n)]
        #pragma unroll
        for (int vi = 0; vi < 8; ++vi) {
            const int r = lg * 8 + vi;
            const int i = m0 + r;
            const int b = i >> 11, n = i & (SEQ - 1);
            #pragma unroll
            for (int t = 0; t < 4; ++t) {
                const int d = t * 16 + ln;
                dst16[((size_t)(b * NHEADS + h) * DHEAD + d) * SEQ + n] =
                    (_Float16)acc[t][vi];
            }
        }
    } else {
        // final projection: fp32 + bias
        #pragma unroll
        for (int vi = 0; vi < 8; ++vi) {
            const int r = lg * 8 + vi;
            const int i = m0 + r;
            #pragma unroll
            for (int t = 0; t < 4; ++t) {
                const int col = n0 + t * 16 + ln;
                dst32[(size_t)i * DIMX + col] = acc[t][vi] + aux[col];
            }
        }
    }
}

// ---------------- flash attention: 4 waves/block, 16 queries/wave -----------
// Waves in a block work on the same (b,h) => K/V tile loads are shared in
// the WGP caches.  Per-wave LDS slice for the P (C->A layout) bounce.
__global__ __launch_bounds__(128) void attn_kernel(
        const _Float16* __restrict__ qn, const _Float16* __restrict__ kn,
        const _Float16* __restrict__ vT, _Float16* __restrict__ aout) {
    const int lane = threadIdx.x & 31;
    const int wv   = threadIdx.x >> 5;
    const int ln = lane & 15, lg = lane >> 4;
    const int bh  = blockIdx.x >> 5;               // b*NHEADS + h
    const int mb  = (blockIdx.x & 31) * 4 + wv;    // query block within (b,h)
    const int m0  = mb * 16;

    const _Float16* q = qn + (size_t)bh * SEQ * DHEAD;
    const _Float16* k = kn + (size_t)bh * SEQ * DHEAD;
    const _Float16* v = vT + (size_t)bh * DHEAD * SEQ;

    const v16h qa0 = load_a_nt(q, DHEAD, m0, 0);
    const v16h qa1 = load_a_nt(q, DHEAD, m0, 32);

    v8f oacc[4] = {};
    float mrow[8], lrow[8];
    #pragma unroll
    for (int vi = 0; vi < 8; ++vi) { mrow[vi] = -INFINITY; lrow[vi] = 0.f; }

    __shared__ _Float16 pl[4][16 * 32];   // per-wave P tile bounce
    _Float16* plw = pl[wv];

    for (int j0 = 0; j0 < SEQ; j0 += 32) {
        // S = Q K^T for 32 keys: two 16x16 C tiles, K-dim = 64 (2 steps)
        v8f s0 = {}, s1 = {};
        s0 = wmma32(qa0, load_b_nt(k, DHEAD, j0, 0), s0);
        s0 = wmma32(qa1, load_b_nt(k, DHEAD, j0, 32), s0);
        s1 = wmma32(qa0, load_b_nt(k, DHEAD, j0 + 16, 0), s1);
        s1 = wmma32(qa1, load_b_nt(k, DHEAD, j0 + 16, 32), s1);

        // online softmax; row r = lg*8+vi lives in VGPR vi of lane group lg
        #pragma unroll
        for (int vi = 0; vi < 8; ++vi) {
            float mx = fmaxf(s0[vi], s1[vi]);
            mx = fmaxf(mx, __shfl_xor(mx, 1, 32));
            mx = fmaxf(mx, __shfl_xor(mx, 2, 32));
            mx = fmaxf(mx, __shfl_xor(mx, 4, 32));
            mx = fmaxf(mx, __shfl_xor(mx, 8, 32));
            const float mnew  = fmaxf(mrow[vi], mx);
            const float alpha = __expf(mrow[vi] - mnew);
            const float p0 = __expf(s0[vi] - mnew);
            const float p1 = __expf(s1[vi] - mnew);
            float psum = p0 + p1;
            psum += __shfl_xor(psum, 1, 32);
            psum += __shfl_xor(psum, 2, 32);
            psum += __shfl_xor(psum, 4, 32);
            psum += __shfl_xor(psum, 8, 32);
            lrow[vi] = lrow[vi] * alpha + psum;
            mrow[vi] = mnew;
            #pragma unroll
            for (int t = 0; t < 4; ++t) oacc[t][vi] *= alpha;
            const int r = lg * 8 + vi;
            plw[r * 32 + ln]      = (_Float16)p0;
            plw[r * 32 + 16 + ln] = (_Float16)p1;
        }
        lds_fence_wave();                      // stores -> visible to wave
        const v16h pa = load_a_nt(plw, 32, 0, 0);
        asm volatile("" ::: "memory");         // keep pa load before reuse
        #pragma unroll
        for (int t = 0; t < 4; ++t)
            oacc[t] = wmma32(pa, load_b_nt(v, SEQ, t * 16, j0), oacc[t]);
    }

    // epilogue: O/l, write (B,N,H*Dh) f16 for the final projection
    const int b = bh >> 4, h = bh & 15;
    #pragma unroll
    for (int vi = 0; vi < 8; ++vi) {
        const int r = lg * 8 + vi;
        const size_t i = (size_t)b * SEQ + m0 + r;
        const float inv = 1.0f / lrow[vi];
        #pragma unroll
        for (int t = 0; t < 4; ++t) {
            const int col = h * DHEAD + t * 16 + ln;
            aout[i * DIMX + col] = (_Float16)(oacc[t][vi] * inv);
        }
    }
}

// ---------------- host-side orchestration ----------------
extern "C" void kernel_launch(void* const* d_in, const int* in_sizes, int n_in,
                              void* d_out, int out_size, void* d_ws, size_t ws_size,
                              hipStream_t stream) {
    const float* x    = (const float*)d_in[0];
    const float* ln_g = (const float*)d_in[1];
    const float* ln_b = (const float*)d_in[2];
    const float* q_g  = (const float*)d_in[3];
    const float* k_g  = (const float*)d_in[4];
    const float* wq   = (const float*)d_in[5];
    const float* wk   = (const float*)d_in[6];
    const float* wv   = (const float*)d_in[7];
    const float* wo   = (const float*)d_in[8];
    const float* bo   = (const float*)d_in[9];
    float* out = (float*)d_out;

    char* ws = (char*)d_ws;
    const size_t MB = (size_t)1 << 20;
    _Float16* xn16 = (_Float16*)(ws + 0);          // 8 MB (reused as aout)
    _Float16* wq16 = (_Float16*)(ws + 8  * MB);    // 2 MB each
    _Float16* wk16 = (_Float16*)(ws + 10 * MB);
    _Float16* wv16 = (_Float16*)(ws + 12 * MB);
    _Float16* wo16 = (_Float16*)(ws + 14 * MB);
    _Float16* qn16 = (_Float16*)(ws + 16 * MB);    // 8 MB each
    _Float16* kn16 = (_Float16*)(ws + 24 * MB);
    _Float16* vT16 = (_Float16*)(ws + 32 * MB);
    _Float16* ao16 = xn16;  // xn dead after QKV projections

    const int n4 = DIMX * DIMX / 4;
    cvt_f16_kernel<<<(n4 + 255) / 256, 256, 0, stream>>>(wq, wq16, n4);
    cvt_f16_kernel<<<(n4 + 255) / 256, 256, 0, stream>>>(wk, wk16, n4);
    cvt_f16_kernel<<<(n4 + 255) / 256, 256, 0, stream>>>(wv, wv16, n4);
    cvt_f16_kernel<<<(n4 + 255) / 256, 256, 0, stream>>>(wo, wo16, n4);

    ln_kernel<<<ROWS, 256, 0, stream>>>(x, ln_g, ln_b, xn16);

    dim3 gg(ROWS / 64, DIMX / 64);   // 64 x 16 blocks, 4 waves each
    gemm_nt_kernel<<<gg, 128, 0, stream>>>(xn16, wq16, qn16, nullptr, q_g, 1);
    gemm_nt_kernel<<<gg, 128, 0, stream>>>(xn16, wk16, kn16, nullptr, k_g, 2);
    gemm_nt_kernel<<<gg, 128, 0, stream>>>(xn16, wv16, vT16, nullptr, nullptr, 0);

    attn_kernel<<<BATCH * NHEADS * (SEQ / 64), 128, 0, stream>>>(
        qn16, kn16, vT16, ao16);

    gemm_nt_kernel<<<gg, 128, 0, stream>>>(ao16, wo16, nullptr, out, bo, 3);
}
